// CustomLSTM_70428873720039
// MI455X (gfx1250) — compile-verified, matrix operand order checked by reading, not run on hardware
//
#include <hip/hip_runtime.h>

// ---------------------------------------------------------------------------
// Types for CDNA5 WMMA (wave32): V_WMMA_F32_16X16X32_BF16
// ---------------------------------------------------------------------------
typedef __bf16 bf16x16 __attribute__((ext_vector_type(16)));
typedef float  v8f     __attribute__((ext_vector_type(8)));

#define EMB 512
#define HID 512
#define BATCH 64
#define SEQ 1024
#define BT (BATCH * SEQ)          // 65536 rows
#define GATES 2048                // 4*HID
#define NSLICE 8                  // gate-dim tensor parallelism in the scan
#define SLICE_N 256               // gate rows per slice (4 gates x 64 h)
#define BBLK 16                   // batch rows per scan group

// K mapping of a 16-bit A/B WMMA fragment (ISA 7.12.2):
//   lane group g = lane>>4 ; element e in v16bf ;  K = (e<8 ? e : e+8) + g*8
__device__ __host__ inline int frag_k(int e, int g) { return (e < 8 ? e : e + 8) + g * 8; }

// ---------------------------------------------------------------------------
// CDNA5 async global->LDS helpers (ASYNCcnt-tracked, per-lane LDS address)
// ---------------------------------------------------------------------------
__device__ inline unsigned lds_off_of(const void* p) {
  // generic -> addrspace(3) -> 32-bit LDS byte offset
  return (unsigned)(size_t)(__attribute__((address_space(3))) const char*)p;
}

__device__ inline void async_ld_b128(unsigned lds_byte_off, const void* gptr) {
  asm volatile("global_load_async_to_lds_b128 %0, %1, off"
               :: "v"(lds_byte_off), "v"((unsigned long long)(size_t)gptr)
               : "memory");
}

__device__ inline void wait_async0() {
#if __has_builtin(__builtin_amdgcn_s_wait_asynccnt)
  __builtin_amdgcn_s_wait_asynccnt(0);
#else
  asm volatile("s_wait_asynccnt 0x0" ::: "memory");
#endif
}

// ---------------------------------------------------------------------------
// Workspace layout (bytes)
// ---------------------------------------------------------------------------
#define XG_BYTES   ((size_t)2 * BT * GATES * 4)          // 1 GB   fp32 xg (both dirs)
#define XBF_BYTES  ((size_t)BT * EMB * 2)                // 64 MB  bf16 inputs
#define WIH_BYTES  ((size_t)2 * GATES * EMB * 2)         // 4 MB   packed bf16 W_ih
#define WHH_BYTES  ((size_t)2 * GATES * HID * 2)         // 4 MB   packed bf16 W_hh
#define BIAS_BYTES ((size_t)2 * GATES * 4)
#define HSH_BYTES  ((size_t)8 * BBLK * HID * 2)          // h exchange (8 groups)
#define BAR_INTS   (8 * 16)

// ---------------------------------------------------------------------------
// fp32 -> bf16 input conversion (streaming, non-temporal)
// ---------------------------------------------------------------------------
__global__ void conv_x_kernel(const float* __restrict__ x, __bf16* __restrict__ xbf, int n) {
  int i = blockIdx.x * 256 + threadIdx.x;
  if (i < n) {
    float v = __builtin_nontemporal_load(x + i);
    xbf[i] = (__bf16)v;
  }
}

__global__ void init_bars_kernel(int* __restrict__ bars) {
  int i = threadIdx.x;
  if (i < BAR_INTS) bars[i] = 0;
}

// ---------------------------------------------------------------------------
// Pack W_ih into WMMA-B-fragment-major bf16, with gate-column permutation:
//   permuted column C = s*256 + q*64 + jj  <->  gate row g = q*512 + s*64 + jj
// Layout: [dir][ct=128][kt=16][lane=32][e=16].  Also emits permuted bias.
// ---------------------------------------------------------------------------
__global__ void pack_wih_kernel(const float* __restrict__ fW, const float* __restrict__ bW,
                                const float* __restrict__ fbi, const float* __restrict__ fbh,
                                const float* __restrict__ bbi, const float* __restrict__ bbh,
                                __bf16* __restrict__ wp, float* __restrict__ biasp) {
  int i = blockIdx.x * 256 + threadIdx.x;
  if (i >= 2 * GATES * EMB) return;
  int e = i & 15, lane = (i >> 4) & 31, kt = (i >> 9) & 15, ct = (i >> 13) & 127, dir = i >> 20;
  int C  = ct * 16 + (lane & 15);
  int s  = C >> 8, rem = C & 255, q = rem >> 6, jj = rem & 63;
  int g  = q * 512 + s * 64 + jj;
  int k  = kt * 32 + frag_k(e, lane >> 4);
  const float* W = dir ? bW : fW;
  wp[i] = (__bf16)W[g * EMB + k];
  if (e == 0 && kt == 0 && lane < 16) {
    biasp[dir * GATES + C] = dir ? (bbi[g] + bbh[g]) : (fbi[g] + fbh[g]);
  }
}

// ---------------------------------------------------------------------------
// Pack W_hh slices, B-fragment-major: [dir][slice=8][kt=16][ntl=16][lane=32][e=16]
// ---------------------------------------------------------------------------
__global__ void pack_whh_kernel(const float* __restrict__ fW, const float* __restrict__ bW,
                                __bf16* __restrict__ wp) {
  int i = blockIdx.x * 256 + threadIdx.x;
  if (i >= 2 * GATES * HID) return;
  int e = i & 15, lane = (i >> 4) & 31, ntl = (i >> 9) & 15, kt = (i >> 13) & 15,
      s = (i >> 17) & 7, dir = i >> 20;
  int Cl = ntl * 16 + (lane & 15);
  int q = Cl >> 6, jj = Cl & 63;
  int g = q * 512 + s * 64 + jj;
  int k = kt * 32 + frag_k(e, lane >> 4);
  wp[i] = (__bf16)(dir ? bW : fW)[g * HID + k];
}

// ---------------------------------------------------------------------------
// Phase 2: xg[dir][bt][C] = x[bt]·W_ihᵀ (permuted cols) + bias, bf16 WMMA.
// Block tile 128x128, K=512. B slab async-loaded to LDS once; A tiles are
// async-copied global->LDS (fragment-major) with double buffering so the
// copy of k+1 overlaps the WMMAs of k.
// ---------------------------------------------------------------------------
__global__ __launch_bounds__(256) void xgemm_kernel(const __bf16* __restrict__ xbf,
                                                    const __bf16* __restrict__ wihp,
                                                    const float* __restrict__ biasp,
                                                    float* __restrict__ xg) {
  __shared__ __bf16 Bs[8 * 16 * 32 * 16];    // [nt=8][kt=16][lane=32][e=16] = 128 KB
  __shared__ __bf16 As[2][8 * 32 * 16];      // double-buffered A fragments  = 16 KB
  const int tid  = threadIdx.x;
  const int mblk = blockIdx.x;   // 0..511
  const int nblk = blockIdx.y;   // 0..15
  const int dir  = blockIdx.z;   // 0..1

  const int wave = tid >> 5, lane = tid & 31;
  const int m  = tid >> 1;            // A row handled by this thread (0..127)
  const int kq = (tid & 1) * 2;       // first of two 8-elem chunks in a 32-elem k-row

  // Per-thread fragment-major LDS destinations for its two A chunks.
  const int mt = m >> 4;
  const int l0 = ((kq & 1) << 4) | (m & 15),     e00 = (kq >> 1) << 3;
  const int l1 = (((kq + 1) & 1) << 4) | (m & 15), e01 = ((kq + 1) >> 1) << 3;
  const unsigned aoff0 = (unsigned)(((mt * 32 + l0) * 16 + e00) * 2);
  const unsigned aoff1 = (unsigned)(((mt * 32 + l1) * 16 + e01) * 2);
  const unsigned abase[2] = { lds_off_of(&As[0][0]), lds_off_of(&As[1][0]) };
  const __bf16* arow = xbf + ((size_t)mblk * 128 + m) * EMB;

  // Async-load B slab (8 column-tiles x 8192 bf16 = 128 KB).
  {
    const __bf16* src = wihp + ((size_t)dir * 128 + nblk * 8) * 8192;
    unsigned bbase = lds_off_of(&Bs[0]);
    for (int i = tid; i < 8192; i += 256)
      async_ld_b128(bbase + (unsigned)i * 16u, (const char*)src + (size_t)i * 16u);
  }
  // Async-load A tile for kt = 0 into buffer 0.
  async_ld_b128(abase[0] + aoff0, arow + kq * 8);
  async_ld_b128(abase[0] + aoff1, arow + (kq + 1) * 8);
  wait_async0();
  __syncthreads();

  v8f acc[8];
  for (int nt = 0; nt < 8; ++nt) acc[nt] = (v8f)0.f;

  for (int kt = 0; kt < 16; ++kt) {
    if (kt < 15) {  // overlap next A-tile copy with this tile's WMMAs
      const __bf16* g = arow + (kt + 1) * 32;
      int nb = (kt + 1) & 1;
      async_ld_b128(abase[nb] + aoff0, g + kq * 8);
      async_ld_b128(abase[nb] + aoff1, g + (kq + 1) * 8);
    }
    bf16x16 a = *(const bf16x16*)&As[kt & 1][(wave * 32 + lane) * 16];
    for (int nt = 0; nt < 8; ++nt) {
      bf16x16 b = *(const bf16x16*)&Bs[((nt * 16 + kt) * 32 + lane) * 16];
      acc[nt] = __builtin_amdgcn_wmma_f32_16x16x32_bf16(false, a, false, b,
                                                        (short)0, acc[nt], false, false);
    }
    wait_async0();
    __syncthreads();
  }

  // Store with bias (C-fragment layout), non-temporal: xg is streamed once.
  const int lm = (lane < 16) ? 0 : 8;
  const int ln = lane & 15;
  const size_t row0 = (size_t)mblk * 128 + wave * 16;
  for (int nt = 0; nt < 8; ++nt) {
    int C = nblk * 128 + nt * 16 + ln;
    float bv = biasp[dir * GATES + C];
    for (int i = 0; i < 8; ++i) {
      size_t r = row0 + lm + i;
      __builtin_nontemporal_store(acc[nt][i] + bv,
                                  &xg[((size_t)dir * BT + r) * GATES + C]);
    }
  }
}

// ---------------------------------------------------------------------------
// Phase 3: recurrent scan. 64 persistent WGs: blockIdx = (dir*4+bblk)*8 + s.
// W_hh slice (256 KB, fragment-major) async-loaded once and kept in LDS.
// Per step: gates = xg + h·W_hhᵀ (WMMA), LSTM cell, publish h slice, 8-WG
// atomic barrier, then async all-gather of the full h tile straight into
// A-fragment-major LDS.  Next step's xg tile is prefetched during compute.
// ---------------------------------------------------------------------------
__device__ inline float sigmoidf_(float x) { return 1.f / (1.f + __expf(-x)); }

__global__ __launch_bounds__(256) void lstm_scan_kernel(const float* __restrict__ xg,
                                                        const __bf16* __restrict__ whhp,
                                                        float* __restrict__ hseq,
                                                        float* __restrict__ hx,
                                                        float* __restrict__ cx,
                                                        __bf16* __restrict__ hsh,
                                                        int* __restrict__ bars) {
  __shared__ __bf16 Wf[16 * 16 * 32 * 16];  // 256 KB: [kt][ntl][lane][e]
  __shared__ __bf16 Hf[16 * 32 * 16];       // 16 KB : [kt][lane][e]  (A fragments of h)
  __shared__ float  Gl[BBLK][SLICE_N];      // 16 KB : gates staging
  __shared__ float  Cs[BBLK][64];           // 4 KB  : cell state

  const int tid   = threadIdx.x;
  const int s     = blockIdx.x & 7;
  const int grp   = blockIdx.x >> 3;  // 0..7
  const int dir   = grp >> 2;
  const int bblk  = grp & 3;
  const int wave  = tid >> 5, lane = tid & 31;
  const int ln    = lane & 15, lg = lane >> 4;

  // Async-load this slice's W_hh fragments (resident for all 1024 steps).
  {
    const __bf16* src = whhp + ((size_t)dir * NSLICE + s) * (512 * SLICE_N);
    unsigned wbase = lds_off_of(&Wf[0]);
    for (int i = tid; i < 16384; i += 256)
      async_ld_b128(wbase + (unsigned)i * 16u, (const char*)src + (size_t)i * 16u);
  }
  // h = 0, c = 0
  {
    uint4 z; z.x = z.y = z.z = z.w = 0u;
    for (int i = tid; i < 1024; i += 256) ((uint4*)Hf)[i] = z;
    for (int i = tid; i < BBLK * 64; i += 256) ((float*)Cs)[i] = 0.f;
  }
  wait_async0();
  __syncthreads();

  __bf16* hg = hsh + (size_t)grp * (BBLK * HID);
  int* bar = bars + grp * 16;   // [0]=count, [1]=phase
  const int n0 = wave * 32;
  const int mlo = lg ? 8 : 0;
  const unsigned hfbase = lds_off_of(&Hf[0]);

  for (int step = 0; step < SEQ; ++step) {
    const int t = dir ? (SEQ - 1 - step) : step;

    // acc <- xg (biases already folded in), loaded directly in C-fragment layout;
    // prefetch next step's tile so its latency hides under the WMMA loop.
    v8f acc0, acc1;
    for (int i = 0; i < 8; ++i) {
      int mrow = mlo + i;
      size_t bt = (size_t)(bblk * BBLK + mrow) * SEQ + t;
      const float* xr = xg + ((size_t)dir * BT + bt) * GATES + s * SLICE_N;
      acc0[i] = xr[n0 + ln];
      acc1[i] = xr[n0 + 16 + ln];
      if (step + 1 < SEQ) {
        const float* xn = xr + (dir ? -(int)GATES : (int)GATES);
        __builtin_prefetch(xn + n0 + ln, 0, 1);
      }
    }

    // gates += h · W_hh_sliceᵀ  (16x256 out, K = 512)
    for (int kt = 0; kt < 16; ++kt) {
      bf16x16 a  = *(const bf16x16*)&Hf[(kt * 32 + lane) * 16];
      bf16x16 b0 = *(const bf16x16*)&Wf[((kt * 16 + wave * 2 + 0) * 32 + lane) * 16];
      bf16x16 b1 = *(const bf16x16*)&Wf[((kt * 16 + wave * 2 + 1) * 32 + lane) * 16];
      acc0 = __builtin_amdgcn_wmma_f32_16x16x32_bf16(false, a, false, b0, (short)0, acc0, false, false);
      acc1 = __builtin_amdgcn_wmma_f32_16x16x32_bf16(false, a, false, b1, (short)0, acc1, false, false);
    }

    // Stage gates to LDS for the cross-lane elementwise phase.
    for (int i = 0; i < 8; ++i) {
      int mrow = mlo + i;
      Gl[mrow][n0 + ln]      = acc0[i];
      Gl[mrow][n0 + 16 + ln] = acc1[i];
    }
    __syncthreads();

    // Elementwise LSTM cell for our 64 h-elements x 16 batch rows.
    {
      int m = tid >> 4;
      int jb = (tid & 15) * 4;
      for (int u = 0; u < 4; ++u) {
        int jj = jb + u;
        float iv = sigmoidf_(Gl[m][jj]);
        float fv = sigmoidf_(Gl[m][64 + jj]);
        float gv = tanhf(Gl[m][128 + jj]);
        float ov = sigmoidf_(Gl[m][192 + jj]);
        float c  = fv * Cs[m][jj] + iv * gv;
        Cs[m][jj] = c;
        float h = ov * tanhf(c);
        int b = bblk * BBLK + m;
        int j = s * 64 + jj;
        __builtin_nontemporal_store(h, &hseq[((size_t)b * SEQ + t) * (2 * HID) + dir * HID + j]);
        hg[m * HID + j] = (__bf16)h;   // publish our h slice
        if (step == SEQ - 1) {
          hx[(size_t)b * (2 * HID) + dir * HID + j] = h;
          cx[(size_t)b * (2 * HID) + dir * HID + j] = c;
        }
      }
    }
    __threadfence();
    __syncthreads();

    // 8-WG sense-reversing barrier for this (dir, batch-block) group.
    if (tid == 0) {
      int ph = atomicAdd(&bar[1], 0);
      if (atomicAdd(&bar[0], 1) == NSLICE - 1) {
        bar[0] = 0;
        __threadfence();
        atomicAdd(&bar[1], 1);
      } else {
        while (atomicAdd(&bar[1], 0) == ph) __builtin_amdgcn_s_sleep(1);
      }
    }
    __syncthreads();
    __threadfence();

    // Async all-gather of the full 16x512 h tile into A-fragment-major LDS.
    for (int cid = tid; cid < BBLK * HID / 8; cid += 256) {   // 1024 chunks of 8 bf16
      int m  = cid >> 6;
      int k0 = (cid & 63) * 8;
      int kt = k0 >> 5, r8 = (k0 & 31) >> 3;
      unsigned dst = hfbase +
          (unsigned)(((kt * 32 + ((r8 & 1) << 4) + m) * 16 + ((r8 >> 1) << 3)) * 2);
      async_ld_b128(dst, hg + m * HID + k0);
    }
    wait_async0();
    __syncthreads();
  }
}

// ---------------------------------------------------------------------------
// Host launcher
// ---------------------------------------------------------------------------
extern "C" void kernel_launch(void* const* d_in, const int* in_sizes, int n_in,
                              void* d_out, int out_size, void* d_ws, size_t ws_size,
                              hipStream_t stream) {
  (void)in_sizes; (void)n_in; (void)out_size; (void)ws_size;
  const float* x    = (const float*)d_in[0];
  const float* fWih = (const float*)d_in[1];
  const float* fWhh = (const float*)d_in[2];
  const float* fbi  = (const float*)d_in[3];
  const float* fbh  = (const float*)d_in[4];
  const float* bWih = (const float*)d_in[5];
  const float* bWhh = (const float*)d_in[6];
  const float* bbi  = (const float*)d_in[7];
  const float* bbh  = (const float*)d_in[8];

  char* ws = (char*)d_ws;
  float*  xg    = (float*)ws;                                    ws += XG_BYTES;
  __bf16* xbf   = (__bf16*)ws;                                   ws += XBF_BYTES;
  __bf16* wihp  = (__bf16*)ws;                                   ws += WIH_BYTES;
  __bf16* whhp  = (__bf16*)ws;                                   ws += WHH_BYTES;
  float*  biasp = (float*)ws;                                    ws += BIAS_BYTES;
  __bf16* hsh   = (__bf16*)ws;                                   ws += HSH_BYTES;
  int*    bars  = (int*)ws;

  float* out  = (float*)d_out;
  float* hseq = out;                              // 64*1024*1024
  float* hx   = out + (size_t)BATCH * SEQ * 2 * HID;
  float* cx   = hx + (size_t)BATCH * 2 * HID;

  init_bars_kernel<<<1, 128, 0, stream>>>(bars);
  conv_x_kernel<<<(BT * EMB) / 256, 256, 0, stream>>>(x, xbf, BT * EMB);
  pack_wih_kernel<<<(2 * GATES * EMB) / 256, 256, 0, stream>>>(fWih, bWih, fbi, fbh, bbi, bbh,
                                                               wihp, biasp);
  pack_whh_kernel<<<(2 * GATES * HID) / 256, 256, 0, stream>>>(fWhh, bWhh, whhp);
  xgemm_kernel<<<dim3(BT / 128, GATES / 128, 2), 256, 0, stream>>>(xbf, wihp, biasp, xg);
  lstm_scan_kernel<<<64, 256, 0, stream>>>(xg, whhp, hseq, hx, cx, hsh, bars);
}
